// Block_47098611368060
// MI455X (gfx1250) — matrix operand order repaired and checked by reading, not compiled, not run on hardware
//
#include <hip/hip_runtime.h>
#include <hip/hip_bf16.h>

// ---------------------------------------------------------------------------
// Types for WMMA (gfx1250: V_WMMA_F32_16X16X32_BF16, wave32)
// ---------------------------------------------------------------------------
typedef __attribute__((ext_vector_type(16))) __bf16       v16bf;
typedef __attribute__((ext_vector_type(8)))  float        v8f;
typedef __attribute__((ext_vector_type(4)))  unsigned int u32x4;
typedef __attribute__((ext_vector_type(2)))  unsigned int u32x2;
typedef __attribute__((ext_vector_type(4)))  float        f32x4;

union Frag {
    v16bf          v;
    u32x4          q[2];
    unsigned short s[16];
};

__device__ __forceinline__ unsigned short f2bf(float f) {
    union { float f; unsigned u; } c; c.f = f;
    unsigned r = c.u + 0x7FFFu + ((c.u >> 16) & 1u);   // round-to-nearest-even
    return (unsigned short)(r >> 16);
}

__device__ __forceinline__ v8f wmma_bf16(const v16bf& a, const v16bf& b, const v8f& c) {
    // (neg_a, A, neg_b, B, c_mod, C, reuse_a, reuse_b)
    return __builtin_amdgcn_wmma_f32_16x16x32_bf16(false, a, false, b, (short)0, c, false, false);
}

#define GEMM_BIAS   1
#define GEMM_GELU   2
#define GEMM_RES    4
#define GEMM_SPLITV 8

// ---------------------------------------------------------------------------
// Tiled transpose + fp32->bf16 convert:  w[K][N] fp32  ->  wt[N][K] bf16
// 32x32 tiles via LDS, fully coalesced on both sides.
// ---------------------------------------------------------------------------
__global__ __launch_bounds__(256) void cvt_t_kernel(const float* __restrict__ w,
                                                    unsigned short* __restrict__ wt,
                                                    int K, int N) {
    __shared__ unsigned short T[32][33];
    int n0 = blockIdx.x * 32, k0 = blockIdx.y * 32;
    int tx = threadIdx.x & 7, ty = threadIdx.x >> 3;   // 8 x 32
    f32x4 f = *reinterpret_cast<const f32x4*>(w + (size_t)(k0 + ty) * N + n0 + tx * 4);
#pragma unroll
    for (int i = 0; i < 4; ++i) T[ty][tx * 4 + i] = f2bf(f[i]);
    __syncthreads();
    unsigned short o[4];
#pragma unroll
    for (int i = 0; i < 4; ++i) o[i] = T[tx * 4 + i][ty];
    u32x2 p;
    p[0] = (unsigned)o[0] | ((unsigned)o[1] << 16);
    p[1] = (unsigned)o[2] | ((unsigned)o[3] << 16);
    *reinterpret_cast<u32x2*>(wt + (size_t)(n0 + ty) * K + k0 + tx * 4) = p;
}

// ---------------------------------------------------------------------------
// LayerNorm (D == 1024, 256 threads, one row per block) + bf16 cast
// ---------------------------------------------------------------------------
__global__ __launch_bounds__(256) void ln_bf16_kernel(
    const float* __restrict__ x, const float* __restrict__ g,
    const float* __restrict__ bia, unsigned short* __restrict__ y, int D) {
    int row = blockIdx.x;
    int tid = threadIdx.x;
    const float* xr = x + (size_t)row * D;
    float v[4], s = 0.f, ss = 0.f;
#pragma unroll
    for (int i = 0; i < 4; ++i) {
        v[i] = xr[tid + i * 256];
        s += v[i];
        ss += v[i] * v[i];
    }
#pragma unroll
    for (int off = 16; off; off >>= 1) {
        s  += __shfl_xor(s,  off, 32);
        ss += __shfl_xor(ss, off, 32);
    }
    __shared__ float sm[8], sm2[8];
    if ((tid & 31) == 0) { sm[tid >> 5] = s; sm2[tid >> 5] = ss; }
    __syncthreads();
    s = 0.f; ss = 0.f;
#pragma unroll
    for (int i = 0; i < 8; ++i) { s += sm[i]; ss += sm2[i]; }
    float mu  = s / (float)D;
    float var = ss / (float)D - mu * mu;
    float r   = rsqrtf(var + 1e-6f);
#pragma unroll
    for (int i = 0; i < 4; ++i) {
        int c = tid + i * 256;
        float o = (v[i] - mu) * r * g[c] + bia[c];
        y[(size_t)row * D + c] = f2bf(o);
    }
}

// ---------------------------------------------------------------------------
// bf16 WMMA GEMM: C[M,N] = A[M,K] * Bt[N,K]^T   (A row-major, Bt = B^T)
// Block tile 128x256, BK=32, 8 waves in 2x4 grid, wave tile 64x64
// (16 WMMAs / wave / K-step). Double-buffered LDS, one barrier per K-step.
// Epilogue: +bias, tanh-GELU, +fp32 residual, fp32/bf16 store; SPLITV
// scatters the V section of QKV into vt[bh][d][key] (per-head transposed).
// ---------------------------------------------------------------------------
#define LDT 40  // 32 + 8 pad (keeps 16B alignment of 8-element chunks)

__global__ __launch_bounds__(256) void gemm_bf16_kernel(
    const unsigned short* __restrict__ A, const unsigned short* __restrict__ Bt,
    const float* __restrict__ bias, const float* __restrict__ res,
    float* __restrict__ outf, unsigned short* __restrict__ outb,
    unsigned short* __restrict__ vt, int S,
    int M, int N, int K, int flags) {
    __shared__ alignas(16) unsigned short As[2][128 * LDT];  // [m][k]
    __shared__ alignas(16) unsigned short Bs[2][256 * LDT];  // [n][k]

    int tid  = threadIdx.x;
    int lane = tid & 31, wv = tid >> 5;
    int half = lane >> 4, lm = lane & 15;
    int wm = wv >> 2, wn = wv & 3;          // 2x4 wave grid
    int m0 = blockIdx.y * 128, n0g = blockIdx.x * 256;

    int arow = tid >> 1;                    // A staging: 2 threads / row
    int akc  = (tid & 1) * 16;
    const unsigned short* aptr = A + (size_t)(m0 + arow) * K + akc;
    const unsigned short* bptr = Bt + (size_t)(n0g + tid) * K;   // 1 thread / n

    v8f acc[4][4];
    v8f vzero = {0.f, 0.f, 0.f, 0.f, 0.f, 0.f, 0.f, 0.f};
#pragma unroll
    for (int i = 0; i < 4; ++i)
#pragma unroll
        for (int j = 0; j < 4; ++j) acc[i][j] = vzero;

    u32x4 ga[2], gb[4];
    // prologue loads (k0 = 0)
    ga[0] = *reinterpret_cast<const u32x4*>(aptr);
    ga[1] = *reinterpret_cast<const u32x4*>(aptr + 8);
#pragma unroll
    for (int c = 0; c < 4; ++c)
        gb[c] = *reinterpret_cast<const u32x4*>(bptr + c * 8);

    int cur = 0;
    for (int k0 = 0; k0 < K; k0 += 32) {
        // stage current tile into LDS[cur] (pure vector stores, no scatters)
        *reinterpret_cast<u32x4*>(&As[cur][arow * LDT + akc])     = ga[0];
        *reinterpret_cast<u32x4*>(&As[cur][arow * LDT + akc + 8]) = ga[1];
#pragma unroll
        for (int c = 0; c < 4; ++c)
            *reinterpret_cast<u32x4*>(&Bs[cur][tid * LDT + c * 8]) = gb[c];
        __syncthreads();

        if (k0 + 32 < K) {  // issue next tile's global loads under the compute
            const unsigned short* ap = aptr + k0 + 32;
            const unsigned short* bp = bptr + k0 + 32;
            ga[0] = *reinterpret_cast<const u32x4*>(ap);
            ga[1] = *reinterpret_cast<const u32x4*>(ap + 8);
#pragma unroll
            for (int c = 0; c < 4; ++c)
                gb[c] = *reinterpret_cast<const u32x4*>(bp + c * 8);
        }

        // A fragments (ISA 16-bit A layout: lanes0-15 K0-7/16-23, lanes16-31 K8-15/24-31)
        Frag af[4];
#pragma unroll
        for (int i = 0; i < 4; ++i) {
            int ar = wm * 64 + i * 16 + lm;
            af[i].q[0] = *reinterpret_cast<const u32x4*>(&As[cur][ar * LDT + half * 8]);
            af[i].q[1] = *reinterpret_cast<const u32x4*>(&As[cur][ar * LDT + half * 8 + 16]);
        }
#pragma unroll
        for (int j = 0; j < 4; ++j) {
            Frag bf;   // B fragment: lane n = column, lanes0-15 K0-7/16-23, lanes16-31 K8-15/24-31
            int bn = wn * 64 + j * 16 + lm;
            bf.q[0] = *reinterpret_cast<const u32x4*>(&Bs[cur][bn * LDT + half * 8]);
            bf.q[1] = *reinterpret_cast<const u32x4*>(&Bs[cur][bn * LDT + half * 8 + 16]);
#pragma unroll
            for (int i = 0; i < 4; ++i) acc[i][j] = wmma_bf16(af[i].v, bf.v, acc[i][j]);
        }
        cur ^= 1;
    }

    // epilogue: C layout = lane n, vgpr r -> row r + 8*half
#pragma unroll
    for (int i = 0; i < 4; ++i) {
#pragma unroll
        for (int j = 0; j < 4; ++j) {
            int ng  = n0g + wn * 64 + j * 16 + lm;
            float bv = (flags & GEMM_BIAS) ? bias[ng] : 0.f;
#pragma unroll
            for (int r = 0; r < 8; ++r) {
                int mg = m0 + wm * 64 + i * 16 + r + half * 8;
                float vv = acc[i][j][r] + bv;
                if (flags & GEMM_GELU) {
                    float t = 0.7978845608028654f * (vv + 0.044715f * vv * vv * vv);
                    vv = 0.5f * vv * (1.f + tanhf(t));
                }
                if (flags & GEMM_RES) vv += res[(size_t)mg * N + ng];
                if (outf) outf[(size_t)mg * N + ng] = vv;
                if (flags & GEMM_SPLITV) {
                    if (ng < 2048) {
                        outb[(size_t)mg * N + ng] = f2bf(vv);
                    } else {  // V section -> per-head transposed vt[bh][d][key]
                        int hh = (ng - 2048) >> 6, dd = (ng - 2048) & 63;
                        int bb = mg / S, key = mg % S;
                        vt[(((size_t)bb * 16 + hh) * 64 + dd) * S + key] = f2bf(vv);
                    }
                } else if (outb) {
                    outb[(size_t)mg * N + ng] = f2bf(vv);
                }
            }
        }
    }
}

// ---------------------------------------------------------------------------
// Causal flash attention, bf16 WMMA, BARRIER-FREE.
// grid=(B*16 heads, S/128), 256 threads; each wave owns 16 query rows.
// qkv: [B,S,3072] bf16 (q | k | unused). vt: [B*16][64][S] bf16 (V^T per head).
// Q/K/V fragments load straight from global; only the wave-private P tile
// round-trips LDS (same-wave DS ordering is in-order, no syncs needed).
// ---------------------------------------------------------------------------
__global__ __launch_bounds__(256) void attn_kernel(
    const unsigned short* __restrict__ qkv, const unsigned short* __restrict__ vt,
    unsigned short* __restrict__ out, int S) {
    const int D3 = 3072, D = 1024, HD = 64;
    const float scale = 0.125f;  // 1/sqrt(64)

    int bh = blockIdx.x;
    int b = bh >> 4, h = bh & 15;
    int qb = blockIdx.y;
    int tid = threadIdx.x;
    int w = tid >> 5, lane = tid & 31, half = lane >> 4, lm = lane & 15;
    int q0w = qb * 128 + w * 16;

    __shared__ alignas(16) unsigned short Pt[8 * 16 * 72];   // per-wave P tiles
    unsigned short* PtW = &Pt[w * 16 * 72];

    const unsigned short* qbase = qkv + (size_t)b * S * D3;
    const unsigned short* vth   = vt + (size_t)bh * 64 * S;

    // Q fragments: A layout is contiguous 8-element chunks in head-dim
    Frag qf[2];
    {
        const unsigned short* qp = qbase + (size_t)(q0w + lm) * D3 + h * HD;
#pragma unroll
        for (int ks = 0; ks < 2; ++ks) {
            int c0 = ks * 32 + half * 8;
            qf[ks].q[0] = *reinterpret_cast<const u32x4*>(qp + c0);
            qf[ks].q[1] = *reinterpret_cast<const u32x4*>(qp + c0 + 16);
        }
    }

    v8f vzero = {0.f, 0.f, 0.f, 0.f, 0.f, 0.f, 0.f, 0.f};
    v8f acc_o[4];
#pragma unroll
    for (int i = 0; i < 4; ++i) acc_o[i] = vzero;
    float m_i[8], l_i[8];
#pragma unroll
    for (int r = 0; r < 8; ++r) { m_i[r] = -3.0e38f; l_i[r] = 0.f; }

    int nkb = (qb + 1) * 2;  // causal: keys < (qb+1)*128, blocks of 64
    for (int kb = 0; kb < nkb; ++kb) {
        int key0 = kb * 64;

        // ---- S = Q * K^T (K row-major is already B-fragment friendly) ----
        v8f acc_s[4];
#pragma unroll
        for (int nt = 0; nt < 4; ++nt) acc_s[nt] = vzero;
#pragma unroll
        for (int nt = 0; nt < 4; ++nt) {
#pragma unroll
            for (int ks = 0; ks < 2; ++ks) {
                Frag kf;
                const unsigned short* kp = qbase +
                    (size_t)(key0 + nt * 16 + lm) * D3 + 1024 + h * HD +
                    ks * 32 + half * 8;
                kf.q[0] = *reinterpret_cast<const u32x4*>(kp);
                kf.q[1] = *reinterpret_cast<const u32x4*>(kp + 16);
                acc_s[nt] = wmma_bf16(qf[ks].v, kf.v, acc_s[nt]);
            }
        }

        // ---- scale + causal mask + online softmax (per row r, half-wave) ----
#pragma unroll
        for (int r = 0; r < 8; ++r) {
            int m = q0w + r + half * 8;
            float rmax = -1e30f;
#pragma unroll
            for (int nt = 0; nt < 4; ++nt) {
                int n = key0 + nt * 16 + lm;
                float sv = acc_s[nt][r] * scale;
                sv = (n <= m) ? sv : -1e30f;
                acc_s[nt][r] = sv;
                rmax = fmaxf(rmax, sv);
            }
#pragma unroll
            for (int off = 8; off; off >>= 1)
                rmax = fmaxf(rmax, __shfl_xor(rmax, off, 32));
            float mnew = fmaxf(m_i[r], rmax);
            float corr = __expf(m_i[r] - mnew);
            float psum = 0.f;
#pragma unroll
            for (int nt = 0; nt < 4; ++nt) {
                float p = __expf(acc_s[nt][r] - mnew);
                psum += p;
                PtW[(r + half * 8) * 72 + nt * 16 + lm] = f2bf(p);
            }
#pragma unroll
            for (int off = 8; off; off >>= 1) psum += __shfl_xor(psum, off, 32);
            l_i[r] = l_i[r] * corr + psum;
            m_i[r] = mnew;
#pragma unroll
            for (int nt = 0; nt < 4; ++nt) acc_o[nt][r] *= corr;
        }

        // ---- O += P * V (A from wave-private PtW, B straight from vt) ----
#pragma unroll
        for (int ks = 0; ks < 2; ++ks) {
            Frag pa;
            int c0 = ks * 32 + half * 8;
            pa.q[0] = *reinterpret_cast<const u32x4*>(&PtW[lm * 72 + c0]);
            pa.q[1] = *reinterpret_cast<const u32x4*>(&PtW[lm * 72 + c0 + 16]);
#pragma unroll
            for (int nt = 0; nt < 4; ++nt) {
                Frag pv;
                const unsigned short* vp =
                    vth + (size_t)(nt * 16 + lm) * S + key0 + ks * 32 + half * 8;
                pv.q[0] = *reinterpret_cast<const u32x4*>(vp);
                pv.q[1] = *reinterpret_cast<const u32x4*>(vp + 16);
                acc_o[nt] = wmma_bf16(pa.v, pv.v, acc_o[nt]);
            }
        }
    }

    // ---- normalize and store bf16 ----
#pragma unroll
    for (int r = 0; r < 8; ++r) {
        float inv = 1.0f / l_i[r];
        size_t orow = ((size_t)b * S + q0w + r + half * 8) * D + h * HD;
#pragma unroll
        for (int nt = 0; nt < 4; ++nt)
            out[orow + nt * 16 + lm] = f2bf(acc_o[nt][r] * inv);
    }
}

// ---------------------------------------------------------------------------
// Launch: cvt^T weights -> LN1 -> QKV gemm(split V^T) -> attention ->
//         proj(+res) -> LN2 -> MLP1(gelu) -> MLP2(+res)
// ---------------------------------------------------------------------------
extern "C" void kernel_launch(void* const* d_in, const int* in_sizes, int n_in,
                              void* d_out, int out_size, void* d_ws, size_t ws_size,
                              hipStream_t stream) {
    (void)in_sizes; (void)n_in; (void)out_size; (void)ws_size;
    const float* x      = (const float*)d_in[0];
    const float* ln1_s  = (const float*)d_in[1];
    const float* ln1_b  = (const float*)d_in[2];
    const float* w_qkv  = (const float*)d_in[3];
    const float* w_proj = (const float*)d_in[4];
    const float* b_proj = (const float*)d_in[5];
    const float* ln2_s  = (const float*)d_in[6];
    const float* ln2_b  = (const float*)d_in[7];
    const float* w1     = (const float*)d_in[8];
    const float* b1     = (const float*)d_in[9];
    const float* w2     = (const float*)d_in[10];
    const float* b2     = (const float*)d_in[11];
    float* out = (float*)d_out;

    const int Bb = 4, S = 2048, D = 1024, H = 4096;
    const int BS = Bb * S;  // 8192 rows

    char* ws = (char*)d_ws;
    size_t o = 0;
    auto take = [&](size_t bytes) -> void* {
        void* p = ws + o;
        o += (bytes + 255) & ~(size_t)255;
        return p;
    };
    unsigned short* wqkv_t  = (unsigned short*)take((size_t)3 * D * D * 2);  // [3D][D]
    unsigned short* wproj_t = (unsigned short*)take((size_t)D * D * 2);      // [D][D]
    unsigned short* w1_t    = (unsigned short*)take((size_t)H * D * 2);      // [H][D]
    unsigned short* w2_t    = (unsigned short*)take((size_t)D * H * 2);      // [D][H]
    unsigned short* y_bf    = (unsigned short*)take((size_t)BS * D * 2);
    unsigned short* big_bf  = (unsigned short*)take((size_t)BS * H * 2);     // qkv / hidden
    unsigned short* attn_bf = (unsigned short*)take((size_t)BS * D * 2);
    unsigned short* vt      = (unsigned short*)take((size_t)Bb * 16 * 64 * S * 2);
    float*          x1      = (float*)take((size_t)BS * D * 4);

    dim3 blk(256);

    // weight transpose+convert: w[K][N] -> wt[N][K] bf16
    cvt_t_kernel<<<dim3(3 * D / 32, D / 32), blk, 0, stream>>>(w_qkv,  wqkv_t,  D, 3 * D);
    cvt_t_kernel<<<dim3(D / 32,     D / 32), blk, 0, stream>>>(w_proj, wproj_t, D, D);
    cvt_t_kernel<<<dim3(H / 32,     D / 32), blk, 0, stream>>>(w1,     w1_t,    D, H);
    cvt_t_kernel<<<dim3(D / 32,     H / 32), blk, 0, stream>>>(w2,     w2_t,    H, D);

    ln_bf16_kernel<<<BS, blk, 0, stream>>>(x, ln1_s, ln1_b, y_bf, D);

    gemm_bf16_kernel<<<dim3(3 * D / 256, BS / 128), blk, 0, stream>>>(
        y_bf, wqkv_t, nullptr, nullptr, nullptr, big_bf, vt, S,
        BS, 3 * D, D, GEMM_SPLITV);

    attn_kernel<<<dim3(Bb * 16, S / 128), blk, 0, stream>>>(big_bf, vt, attn_bf, S);

    gemm_bf16_kernel<<<dim3(D / 256, BS / 128), blk, 0, stream>>>(
        attn_bf, wproj_t, b_proj, x, x1, nullptr, nullptr, S,
        BS, D, D, GEMM_BIAS | GEMM_RES);

    ln_bf16_kernel<<<BS, blk, 0, stream>>>(x1, ln2_s, ln2_b, y_bf, D);

    gemm_bf16_kernel<<<dim3(H / 256, BS / 128), blk, 0, stream>>>(
        y_bf, w1_t, b1, nullptr, nullptr, big_bf, nullptr, S,
        BS, H, D, GEMM_BIAS | GEMM_GELU);

    gemm_bf16_kernel<<<dim3(D / 256, BS / 128), blk, 0, stream>>>(
        big_bf, w2_t, b2, x1, out, nullptr, nullptr, S,
        BS, D, H, GEMM_BIAS | GEMM_RES);
}
